// AttentionHead_66331474920230
// MI455X (gfx1250) — matrix operand order repaired and checked by reading, not compiled, not run on hardware
//
#include <hip/hip_runtime.h>

#define EMB  1024
#define KDIM 64
#define BB   8
#define SS   2048

typedef __attribute__((ext_vector_type(16))) _Float16 v16h;
typedef __attribute__((ext_vector_type(8)))  _Float16 v8h;
typedef __attribute__((ext_vector_type(8)))  float    v8f;
typedef __attribute__((ext_vector_type(4)))  float    v4f;

__device__ __forceinline__ v16h cat8(v8h lo, v8h hi) {
  v16h r;
#pragma unroll
  for (int i = 0; i < 8; ++i) { r[i] = lo[i]; r[i + 8] = hi[i]; }
  return r;
}

__device__ __forceinline__ v8h cvt8(v4f a, v4f b) {
  v8h r;
#pragma unroll
  for (int i = 0; i < 4; ++i) { r[i] = (_Float16)a[i]; r[i + 4] = (_Float16)b[i]; }
  return r;
}

// ---------------------------------------------------------------------------
// Kernel 0: one-shot weight transpose + f32->f16 convert.
// wq/wk/wv1 [1024 k][64 n] -> w*t [64 n][1024 k] f16 (B-fragment order)
// wv2       [64 k][1024 n] -> wv2t [1024 n][64 k] f16
// 65536 threads, one element of each matrix per thread. L2-resident forever.
// ---------------------------------------------------------------------------
__global__ __launch_bounds__(256) void k_wt(const float* __restrict__ wq,
                                            const float* __restrict__ wk,
                                            const float* __restrict__ wv1,
                                            const float* __restrict__ wv2,
                                            _Float16* __restrict__ wqt,
                                            _Float16* __restrict__ wkt,
                                            _Float16* __restrict__ wv1t,
                                            _Float16* __restrict__ wv2t) {
  const int idx = blockIdx.x * 256 + threadIdx.x;  // 0..65535
  const int n = idx >> 10;        // 0..63
  const int k = idx & 1023;       // 0..1023
  wqt [n * EMB + k] = (_Float16)wq [(long)k * KDIM + n];
  wkt [n * EMB + k] = (_Float16)wk [(long)k * KDIM + n];
  wv1t[n * EMB + k] = (_Float16)wv1[(long)k * KDIM + n];
  const int n2 = idx & 1023;      // 0..1023
  const int k2 = idx >> 10;       // 0..63
  wv2t[n2 * KDIM + k2] = (_Float16)wv2[(long)k2 * EMB + n2];
}

// ---------------------------------------------------------------------------
// Kernel 1: Qh = x*w_q, Kh = x*w_k, Th = x*w_v1   (f16 outputs, [B*S, 64])
// block = 256 thr (8 waves), each wave owns a 16-row tile; 128 rows / block.
// LDS-free, barrier-free: B-fragments are direct b128 loads from the
// pre-transposed f16 weights (K contiguous per lane, L2 resident).
// ---------------------------------------------------------------------------
__global__ __launch_bounds__(256) void k_proj(const float* __restrict__ x,
                                              const _Float16* __restrict__ wqt,
                                              const _Float16* __restrict__ wkt,
                                              const _Float16* __restrict__ wv1t,
                                              _Float16* __restrict__ Qh,
                                              _Float16* __restrict__ Kh,
                                              _Float16* __restrict__ Th) {
  const int tid = threadIdx.x;
  const int lane = tid & 31;
  const int wave = tid >> 5;
  const int halfhi = lane >> 4;   // 0: lanes 0-15, 1: lanes 16-31
  const int l16 = lane & 15;
  const long rowbase = (long)blockIdx.x * 128 + wave * 16;
  const _Float16* wt[3] = {wqt, wkt, wv1t};

  v8f acc[3][4];
  v8f zero = {};
#pragma unroll
  for (int o = 0; o < 3; ++o)
#pragma unroll
    for (int t = 0; t < 4; ++t) acc[o][t] = zero;

  for (int kc = 0; kc < EMB / 32; ++kc) {
    const int kbase = kc * 32;

    // A fragment: 16x32 slice of x, per ISA 16-bit A layout
    const float* xp = x + (rowbase + l16) * (long)EMB + kbase + (halfhi ? 8 : 0);
    v8h alo = cvt8(*(const v4f*)xp, *(const v4f*)(xp + 4));
    v8h ahi = cvt8(*(const v4f*)(xp + 16), *(const v4f*)(xp + 20));
    v16h afrag = cat8(alo, ahi);

#pragma unroll
    for (int o = 0; o < 3; ++o) {
#pragma unroll
      for (int t = 0; t < 4; ++t) {
        const _Float16* bp = wt[o] + (t * 16 + l16) * EMB + kbase + halfhi * 8;
        v16h bfrag = cat8(*(const v8h*)bp, *(const v8h*)(bp + 16));
        acc[o][t] = __builtin_amdgcn_wmma_f32_16x16x32_f16(
            false, afrag, false, bfrag, (short)0, acc[o][t], false, false);
      }
    }
  }

  _Float16* outs[3] = {Qh, Kh, Th};
#pragma unroll
  for (int o = 0; o < 3; ++o)
#pragma unroll
    for (int t = 0; t < 4; ++t)
#pragma unroll
      for (int r = 0; r < 8; ++r) {
        long row = rowbase + r + 8 * halfhi;
        outs[o][row * KDIM + t * 16 + l16] = (_Float16)acc[o][t][r];
      }
}

// ---------------------------------------------------------------------------
// Kernel 2: Vt = (Th * w_v2)^T   (f16, stored TRANSPOSED as [B][EMB][S])
// so that k_attn's PV B-fragments have the key dimension contiguous.
// LDS-free, barrier-free: A straight from Th, B from pre-transposed wv2t.
// ---------------------------------------------------------------------------
__global__ __launch_bounds__(256) void k_vproj(const _Float16* __restrict__ Th,
                                               const _Float16* __restrict__ wv2t,
                                               _Float16* __restrict__ Vt) {
  const int tid = threadIdx.x;
  const int lane = tid & 31;
  const int wave = tid >> 5;
  const int halfhi = lane >> 4;
  const int l16 = lane & 15;
  const long r0 = (long)blockIdx.x * 16;     // global token row over B*S
  const long b   = r0 / SS;
  const long tok = r0 % SS;
  const int koff = halfhi * 8;

  const _Float16* ap = Th + (r0 + l16) * KDIM;
  v16h a0 = cat8(*(const v8h*)(ap + koff),      *(const v8h*)(ap + 16 + koff));
  v16h a1 = cat8(*(const v8h*)(ap + 32 + koff), *(const v8h*)(ap + 48 + koff));

  const int colbase = wave * 128;
#pragma unroll
  for (int t = 0; t < 8; ++t) {
    int n = colbase + t * 16 + l16;
    const _Float16* bp = wv2t + n * KDIM + halfhi * 8;
    v16h b0 = cat8(*(const v8h*)bp,        *(const v8h*)(bp + 16));
    v16h b1 = cat8(*(const v8h*)(bp + 32), *(const v8h*)(bp + 48));
    v8f c = {};
    c = __builtin_amdgcn_wmma_f32_16x16x32_f16(false, a0, false, b0, (short)0, c, false, false);
    c = __builtin_amdgcn_wmma_f32_16x16x32_f16(false, a1, false, b1, (short)0, c, false, false);
    // transposed store: Vt[b][n][tok]
#pragma unroll
    for (int r = 0; r < 8; ++r) {
      long tk = tok + r + 8 * halfhi;
      Vt[(b * EMB + n) * SS + tk] = (_Float16)c[r];
    }
  }
}

// ---------------------------------------------------------------------------
// Kernel 3: fused causal flash attention: out = softmax(Q K^T / 8) * V
// grid = (S/16, B); block = 256 thr (8 waves). Block owns 16 queries; wave w
// owns output columns 128w..128w+127. 32 keys per online-softmax step.
// V^T layout in global means K and V fragments are direct b128 global loads:
// no block barriers, LDS only for the wave-local P (C->A) relayout (8 KB).
// ---------------------------------------------------------------------------
__global__ __launch_bounds__(256) void k_attn(const _Float16* __restrict__ Qh,
                                              const _Float16* __restrict__ Kh,
                                              const _Float16* __restrict__ Vt,
                                              float* __restrict__ out) {
  __shared__ _Float16 phs[8][16][32] __attribute__((aligned(64)));  // 8 KB
  const int tid = threadIdx.x;
  const int lane = tid & 31;
  const int wave = tid >> 5;
  const int halfhi = lane >> 4;
  const int l16 = lane & 15;
  const int b = blockIdx.y;
  const int q0 = blockIdx.x * 16;
  const long rowQ = (long)b * SS + q0;
  const int koff = halfhi * 8;
  _Float16* ph = &phs[wave][0][0];

  // Q fragments (16 queries x 64 dims -> 2 k-chunks); fold in 1/sqrt(64)
  const _Float16* qp = Qh + (rowQ + l16) * KDIM;
  v16h aq0 = cat8(*(const v8h*)(qp + koff),      *(const v8h*)(qp + 16 + koff));
  v16h aq1 = cat8(*(const v8h*)(qp + 32 + koff), *(const v8h*)(qp + 48 + koff));
  const _Float16 hscale = (_Float16)0.125f;   // exact power of two
#pragma unroll
  for (int i = 0; i < 16; ++i) { aq0[i] *= hscale; aq1[i] *= hscale; }

  float mrow[8], lrow[8];
  v8f acc[8];
  v8f zero = {};
#pragma unroll
  for (int r = 0; r < 8; ++r) { mrow[r] = -1e30f; lrow[r] = 0.f; }
#pragma unroll
  for (int t = 0; t < 8; ++t) acc[t] = zero;

  const int nsteps = (q0 + 16 + 31) >> 5;
  const long vbase = (long)b * EMB * SS;
  const int cb = wave * 128;

  for (int s = 0; s < nsteps; ++s) {
    const int ktb = s * 32;

    // prefetch next step's K rows (probe-confirmed -> global_prefetch_b8)
    if (s + 1 < nsteps) {
      __builtin_prefetch(Kh + ((long)b * SS + ktb + 32 + lane) * KDIM, 0, 1);
    }

    // ---- scores: 16 queries x 32 keys (2 n-tiles x 2 k-chunks) ----
    v8f c[2];
#pragma unroll
    for (int t = 0; t < 2; ++t) {
      int key = ktb + t * 16 + l16;
      const _Float16* kp = Kh + ((long)b * SS + key) * KDIM + halfhi * 8;
      v16h bk0 = cat8(*(const v8h*)kp,        *(const v8h*)(kp + 16));
      v16h bk1 = cat8(*(const v8h*)(kp + 32), *(const v8h*)(kp + 48));
      c[t] = zero;
      c[t] = __builtin_amdgcn_wmma_f32_16x16x32_f16(false, aq0, false, bk0, (short)0, c[t], false, false);
      c[t] = __builtin_amdgcn_wmma_f32_16x16x32_f16(false, aq1, false, bk1, (short)0, c[t], false, false);
    }

    // ---- causal mask (scale already folded into Q) ----
    float pm[2][8];
#pragma unroll
    for (int t = 0; t < 2; ++t) {
      int key = ktb + t * 16 + l16;
#pragma unroll
      for (int r = 0; r < 8; ++r) {
        int q = q0 + r + 8 * halfhi;
        pm[t][r] = (key <= q) ? c[t][r] : -1e30f;
      }
    }

    // ---- online softmax: row max across 32 keys (16-lane half reductions) ----
    float mnew[8], alpha[8];
#pragma unroll
    for (int r = 0; r < 8; ++r) {
      float v = fmaxf(pm[0][r], pm[1][r]);
#pragma unroll
      for (int off = 1; off < 16; off <<= 1)
        v = fmaxf(v, __shfl_xor(v, off, 32));
      mnew[r] = fmaxf(mrow[r], v);
      alpha[r] = __expf(mrow[r] - mnew[r]);
      mrow[r] = mnew[r];
    }

    // ---- P = exp(S - m), write P tile to wave-local LDS (C -> A relayout) ----
#pragma unroll
    for (int t = 0; t < 2; ++t)
#pragma unroll
      for (int r = 0; r < 8; ++r) {
        float p = __expf(pm[t][r] - mnew[r]);
        pm[t][r] = p;
        ph[(r + 8 * halfhi) * 32 + t * 16 + l16] = (_Float16)p;
      }

#pragma unroll
    for (int r = 0; r < 8; ++r) {
      float rs = pm[0][r] + pm[1][r];
#pragma unroll
      for (int off = 1; off < 16; off <<= 1)
        rs += __shfl_xor(rs, off, 32);
      lrow[r] = lrow[r] * alpha[r] + rs;
    }
#pragma unroll
    for (int t = 0; t < 8; ++t)
#pragma unroll
      for (int r = 0; r < 8; ++r) acc[t][r] *= alpha[r];

    // ---- O += P * V : A from wave-local LDS P, B direct from global V^T ----
    const _Float16* pp = ph + l16 * 32 + koff;
    v16h apv = cat8(*(const v8h*)pp, *(const v8h*)(pp + 16));
#pragma unroll
    for (int t = 0; t < 8; ++t) {
      const _Float16* vp = Vt + (vbase + (long)(cb + t * 16 + l16) * SS) + ktb + halfhi * 8;
      if (s + 1 < nsteps) __builtin_prefetch(vp + 32, 0, 1);
      v16h bv = cat8(*(const v8h*)vp, *(const v8h*)(vp + 16));
      acc[t] = __builtin_amdgcn_wmma_f32_16x16x32_f16(false, apv, false, bv, (short)0, acc[t], false, false);
    }
  }

  // ---- epilogue: normalize (one reciprocal per row) and write fp32 output ----
  float inv[8];
#pragma unroll
  for (int r = 0; r < 8; ++r) inv[r] = 1.0f / lrow[r];
#pragma unroll
  for (int t = 0; t < 8; ++t) {
    int n = cb + t * 16 + l16;
#pragma unroll
    for (int r = 0; r < 8; ++r) {
      long row = rowQ + r + 8 * halfhi;
      out[row * EMB + n] = acc[t][r] * inv[r];
    }
  }
}

// ---------------------------------------------------------------------------
extern "C" void kernel_launch(void* const* d_in, const int* in_sizes, int n_in,
                              void* d_out, int out_size, void* d_ws, size_t ws_size,
                              hipStream_t stream) {
  const float* x   = (const float*)d_in[0];
  const float* wq  = (const float*)d_in[1];
  const float* wk  = (const float*)d_in[2];
  const float* wv1 = (const float*)d_in[3];
  const float* wv2 = (const float*)d_in[4];
  float* out = (float*)d_out;

  char* ws = (char*)d_ws;
  _Float16* Qh   = (_Float16*)(ws + (size_t)0);                  // 2 MB
  _Float16* Kh   = (_Float16*)(ws + (size_t)2 * 1024 * 1024);    // 2 MB
  _Float16* Th   = (_Float16*)(ws + (size_t)4 * 1024 * 1024);    // 2 MB
  _Float16* Vt   = (_Float16*)(ws + (size_t)6 * 1024 * 1024);    // 32 MB (transposed)
  char* wbase    = ws + (size_t)38 * 1024 * 1024;
  _Float16* wqt  = (_Float16*)(wbase);                           // 128 KB each
  _Float16* wkt  = (_Float16*)(wbase + 131072);
  _Float16* wv1t = (_Float16*)(wbase + 2 * 131072);
  _Float16* wv2t = (_Float16*)(wbase + 3 * 131072);

  k_wt   <<<dim3(256),            dim3(256), 0, stream>>>(wq, wk, wv1, wv2, wqt, wkt, wv1t, wv2t);
  k_proj <<<dim3(BB * SS / 128),  dim3(256), 0, stream>>>(x, wqt, wkt, wv1t, Qh, Kh, Th);
  k_vproj<<<dim3(BB * SS / 16),   dim3(256), 0, stream>>>(Th, wv2t, Vt);
  k_attn <<<dim3(SS / 16, BB),    dim3(256), 0, stream>>>(Qh, Kh, Vt, out);
}
